// CausalSelfAttention_86560770883803
// MI455X (gfx1250) — matrix operand order repaired and checked by reading, not compiled, not run on hardware
//
#include <hip/hip_runtime.h>
#include <hip/hip_bf16.h>
#include <math.h>

// ---------------------------------------------------------------------------
// CDNA5 (gfx1250, wave32) causal self-attention forward.
//   B=8, T=1024, C=2048, H=16, D=128.
//   bf16 WMMA (v_wmma_f32_16x16x32_bf16) with f32 accumulation,
//   flash-attention (no TxT materialization),
//   TDM (tensor_load_to_lds) staging, double-buffered GEMM pipeline.
// ---------------------------------------------------------------------------

#define ATTN_USE_TDM 1   // flip to 0 if the tensor_load_to_lds asm is rejected

typedef __attribute__((ext_vector_type(16))) __bf16 v16bf;
typedef __attribute__((ext_vector_type(8)))  __bf16 v8bf;
typedef __attribute__((ext_vector_type(8)))  float  v8f;
typedef __attribute__((ext_vector_type(4)))  unsigned int u32x4;
typedef __attribute__((ext_vector_type(8)))  unsigned int u32x8;

__device__ __forceinline__ unsigned short f32_to_bf16(float f) {
  union { float f; unsigned int u; } v; v.f = f;
  unsigned int u = v.u;
  u += 0x7FFFu + ((u >> 16) & 1u);   // round-to-nearest-even
  return (unsigned short)(u >> 16);
}

// Build a v16bf fragment from two 16-byte-aligned chunks of 8 bf16 each.
__device__ __forceinline__ v16bf frag16(const unsigned short* p0,
                                        const unsigned short* p1) {
  v8bf lo = *(const v8bf*)p0;
  v8bf hi = *(const v8bf*)p1;
  return __builtin_shufflevector(lo, hi, 0,1,2,3,4,5,6,7,8,9,10,11,12,13,14,15);
}

__device__ __forceinline__ void store_out(float* p, float v)          { *p = v; }
__device__ __forceinline__ void store_out(unsigned short* p, float v) { *p = f32_to_bf16(v); }

// Low 32 bits of a generic pointer to a __shared__ object = LDS byte offset.
__device__ __forceinline__ unsigned lds_off(const void* p) {
  return (unsigned)(unsigned long long)p;
}

#if ATTN_USE_TDM
// ---------------------------------------------------------------------------
// TDM: issue a 2D tensor_load_to_lds.  D# per CDNA5 ISA ch.8:
//   group0: count=1, lds_addr, global_addr[56:0], type=2
//   group1: data_size=2B, optional LDS padding, tile dims, row stride.
// Issue from ONE wave only (EXEC ignored; per-wave instruction).
// ---------------------------------------------------------------------------
__device__ __forceinline__ void tdm_load_2d(const void* gptr, unsigned lds_byte,
                                            unsigned tile_d0, unsigned tile_d1,
                                            unsigned long long stride0_elems,
                                            unsigned pad_interval_code,
                                            unsigned pad_amount_code,
                                            bool pad_en) {
  unsigned long long ga = (unsigned long long)gptr;
  u32x4 g0;
  g0[0] = 1u;                                        // count = 1 descriptor
  g0[1] = lds_byte;                                  // lds_addr
  g0[2] = (unsigned)ga;                              // global_addr[31:0]
  g0[3] = (unsigned)((ga >> 32) & 0x01FFFFFFu)       // global_addr[56:32]
        | (2u << 30);                                // type = 2 (image)
  const unsigned long long TD0 = 0x40000000ull;      // huge tensor dims:
  const unsigned long long TD1 = 0x40000000ull;      // no OOB clipping
  unsigned flags = (1u << 16);                       // data_size = 2 bytes
  if (pad_en)
    flags |= (1u << 20) | (pad_interval_code << 22) | (pad_amount_code << 25);
  u32x8 g1;
  g1[0] = flags;
  g1[1] = (unsigned)((TD0 & 0xFFFFull) << 16);
  g1[2] = (unsigned)(((TD0 >> 16) & 0xFFFFull) | ((TD1 & 0xFFFFull) << 16));
  g1[3] = (unsigned)(((TD1 >> 16) & 0xFFFFull) | ((unsigned long long)(tile_d0 & 0xFFFFu) << 16));
  g1[4] = (tile_d1 & 0xFFFFu);                       // tile_dim2 = 0
  g1[5] = (unsigned)(stride0_elems & 0xFFFFFFFFull);
  g1[6] = (unsigned)((stride0_elems >> 32) & 0xFFFFull);   // stride1 = 0
  g1[7] = 0u;
  asm volatile("tensor_load_to_lds %0, %1" :: "s"(g0), "s"(g1) : "memory");
}
__device__ __forceinline__ void tdm_wait() {
  __builtin_amdgcn_s_wait_tensorcnt((short)0);
}
#endif

// ---------------------------------------------------------------------------
// fp32 -> bf16 conversion
// ---------------------------------------------------------------------------
__global__ void cvt_f32_bf16(const float* __restrict__ in,
                             unsigned short* __restrict__ out, size_t n) {
  size_t i = (size_t)blockIdx.x * blockDim.x + threadIdx.x;
  size_t stride = (size_t)gridDim.x * blockDim.x;
  for (; i < n; i += stride) out[i] = f32_to_bf16(in[i]);
}

// ---------------------------------------------------------------------------
// bf16 GEMM:  C[M,N] = A[M,K] * B[K,N], row-major, dims multiples of 128/64.
// 256 threads = 8 waves (2 x 4), 128x128 C tile, BK = 64 (2 WMMA K-steps).
// Double-buffered pipeline: TDM DMA for A tile t+1 and global loads for
// B tile t+1 are issued BEFORE the 16 WMMAs on tile t; s_wait_tensorcnt
// + one barrier per K-step close the stage.
// ---------------------------------------------------------------------------
template <typename OutT>
__global__ __launch_bounds__(256) void gemm_bf16_wmma(
    const unsigned short* __restrict__ A, const unsigned short* __restrict__ B,
    OutT* __restrict__ Cd, int M, int N, int K) {
  __shared__ __align__(16) unsigned short As[2][128][72];  // [m][k], 64+8 pad
  __shared__ __align__(16) unsigned short Bs[2][128][72];  // [n][k] transposed

  const int tid  = threadIdx.x;
  const int wave = tid >> 5, lane = tid & 31;
  const int wm   = wave >> 2, wn   = wave & 3;       // wave grid 2 x 4
  const int lm   = lane & 15, sel  = lane >> 4;      // lane row/col, half-sel

  const int m0 = blockIdx.y * 128;
  const int n0 = blockIdx.x * 128;

  auto stageA = [&](int k0, int buf) {
#if ATTN_USE_TDM
    if (wave == 0)
      tdm_load_2d(A + (size_t)m0 * K + k0, lds_off(&As[buf][0][0]),
                  /*d0*/64, /*d1*/128, (unsigned long long)K,
                  /*interval 128B*/4, /*pad 16B*/3, true);
#else
    const int row = tid >> 1, seg = (tid & 1) * 32;
    const unsigned short* ga = A + (size_t)(m0 + row) * K + k0 + seg;
#pragma unroll
    for (int q = 0; q < 4; ++q)
      *(uint4*)&As[buf][row][seg + q * 8] = *(const uint4*)(ga + q * 8);
#endif
  };
  auto stageB = [&](int k0, int buf) {
    const int n8 = (tid & 15) * 8;                   // 0..120
#pragma unroll
    for (int pass = 0; pass < 2; ++pass) {
      const int k = (tid >> 4) * 2 + pass * 32;      // even k, 0..62
      union { uint4 v; unsigned short s[8]; } u0, u1;
      u0.v = *(const uint4*)(B + (size_t)(k0 + k)     * N + n0 + n8);
      u1.v = *(const uint4*)(B + (size_t)(k0 + k + 1) * N + n0 + n8);
#pragma unroll
      for (int e = 0; e < 8; ++e)
        *(unsigned int*)&Bs[buf][n8 + e][k] =
            (unsigned int)u0.s[e] | ((unsigned int)u1.s[e] << 16);
    }
  };

  const v8f vzero = {0.f,0.f,0.f,0.f,0.f,0.f,0.f,0.f};
  v8f acc[4][2];
#pragma unroll
  for (int mt = 0; mt < 4; ++mt)
#pragma unroll
    for (int nt = 0; nt < 2; ++nt) acc[mt][nt] = vzero;

  // ---- prologue: stage tile 0 ----
  int cur = 0;
  stageA(0, 0);
  stageB(0, 0);
#if ATTN_USE_TDM
  if (wave == 0) tdm_wait();
#endif
  __syncthreads();

  const int NT = K / 64;
  for (int t = 0; t < NT; ++t) {
    const bool more = (t + 1 < NT);
    if (more) {                       // prefetch tile t+1 into the other buffer
      stageA((t + 1) * 64, cur ^ 1);
      stageB((t + 1) * 64, cur ^ 1);
    }
    // ---- compute tile t: 2 WMMA K-steps of 32, 8 WMMAs each ----
#pragma unroll
    for (int kk = 0; kk < 2; ++kk) {
      v16bf af[4], bfv[2];
#pragma unroll
      for (int mt = 0; mt < 4; ++mt) {
        const unsigned short* p = &As[cur][wm * 64 + mt * 16 + lm][kk * 32];
        af[mt] = frag16(p + 16 * sel, p + 16 * sel + 8);
      }
#pragma unroll
      for (int nt = 0; nt < 2; ++nt) {
        const unsigned short* p = &Bs[cur][wn * 32 + nt * 16 + lm][kk * 32 + 16 * sel];
        bfv[nt] = frag16(p, p + 8);
      }
#pragma unroll
      for (int mt = 0; mt < 4; ++mt)
#pragma unroll
        for (int nt = 0; nt < 2; ++nt)
          acc[mt][nt] = __builtin_amdgcn_wmma_f32_16x16x32_bf16(
              false, af[mt], false, bfv[nt], (short)0, acc[mt][nt], false, false);
    }
#if ATTN_USE_TDM
    if (wave == 0 && more) tdm_wait();
#endif
    __syncthreads();
    cur ^= 1;
  }

  // ---- writeback: C/D layout => row = r + 8*sel, col = lm ----
#pragma unroll
  for (int mt = 0; mt < 4; ++mt)
#pragma unroll
    for (int nt = 0; nt < 2; ++nt)
#pragma unroll
      for (int r = 0; r < 8; ++r) {
        const int row = m0 + wm * 64 + mt * 16 + r + 8 * sel;
        const int col = n0 + wn * 32 + nt * 16 + lm;
        store_out(&Cd[(size_t)row * N + col], acc[mt][nt][r]);
      }
}

// ---------------------------------------------------------------------------
// Flash attention: one workgroup (4 waves) = 64 query rows of one (b,h).
// K block staged by TDM (row-major); V transposed by VALU (paired b32);
// online softmax in f32; P restaged via wave-private LDS for P @ V.
// B fragments are preloaded in batches so ds_loads overlap the WMMA chain.
// ---------------------------------------------------------------------------
__global__ __launch_bounds__(128) void attn_flash_wmma(
    const unsigned short* __restrict__ qkv,   // [B,T,3C] bf16
    unsigned short* __restrict__ Y) {         // [B,T,C]  bf16
  const int T = 1024, C = 2048, D = 128, TRC = 3 * C;

  __shared__ __align__(16) unsigned short Kls[64][136];    // [key][d], padded
  __shared__ __align__(16) unsigned short Vt[128][72];     // [d][key], padded
  __shared__ __align__(16) unsigned short Pls[4][16][72];  // per-wave P, padded

  const int tid  = threadIdx.x;
  const int wave = tid >> 5, lane = tid & 31;
  const int lm   = lane & 15, sel  = lane >> 4;
  const int qblk = blockIdx.x;                 // 0..15 (64-row q block)
  const int b    = blockIdx.y >> 4;
  const int h    = blockIdx.y & 15;

  // ---- load Q fragments once (A-layout, 4 fragments cover d = 0..127) ----
  const int qrow = qblk * 64 + wave * 16 + lm;
  const unsigned short* qp = qkv + (size_t)(b * T + qrow) * TRC + h * D;
  v16bf qf[4];
#pragma unroll
  for (int kk = 0; kk < 4; ++kk)
    qf[kk] = frag16(qp + kk * 32 + 16 * sel, qp + kk * 32 + 16 * sel + 8);

  const v8f vzero = {0.f,0.f,0.f,0.f,0.f,0.f,0.f,0.f};
  v8f o[8];
  float mrow[8], lrow[8];
#pragma unroll
  for (int i = 0; i < 8; ++i) { o[i] = vzero; mrow[i] = -1e30f; lrow[i] = 0.f; }

  const float scale = 0.08838834764831845f;    // 1/sqrt(128)

  for (int j = 0; j <= qblk; ++j) {
    const unsigned short* kvbase = qkv + (size_t)(b * T + j * 64) * TRC + h * D;
    // ---- K block: TDM DMA, 64 rows x 128 elems, rows padded 256B+16B ----
#if ATTN_USE_TDM
    if (wave == 0)
      tdm_load_2d(kvbase + C, lds_off(&Kls[0][0]),
                  /*d0*/128, /*d1*/64, (unsigned long long)TRC,
                  /*interval 256B*/5, /*pad 16B*/3, true);
#else
#pragma unroll
    for (int i = 0; i < 4; ++i) {
      const int idx = tid + i * 128;
      const int r = idx >> 3, c8 = (idx & 7) * 16;
      const unsigned short* kp = kvbase + (size_t)r * TRC + C + c8;
      *(uint4*)&Kls[r][c8]     = *(const uint4*)kp;
      *(uint4*)&Kls[r][c8 + 8] = *(const uint4*)(kp + 8);
    }
#endif
    // ---- V block transposed into Vt[d][key]; paired keys -> b32 stores ----
#pragma unroll
    for (int i = 0; i < 4; ++i) {
      const int idx = tid + i * 128;           // 0..511 pair slots
      const int kp2 = (idx >> 4) * 2;          // even key 0..62
      const int c8  = (idx & 15) * 8;          // d chunk
      union { uint4 v; unsigned short s[8]; } u0, u1;
      u0.v = *(const uint4*)(kvbase + (size_t)kp2       * TRC + 2 * C + c8);
      u1.v = *(const uint4*)(kvbase + (size_t)(kp2 + 1) * TRC + 2 * C + c8);
#pragma unroll
      for (int e = 0; e < 8; ++e)
        *(unsigned int*)&Vt[c8 + e][kp2] =
            (unsigned int)u0.s[e] | ((unsigned int)u1.s[e] << 16);
    }
#if ATTN_USE_TDM
    if (wave == 0) tdm_wait();
#endif
    __syncthreads();

    // ---- S = Q * K^T: preload the 4 K-fragments per n-tile, then chain ----
    v8f s[4];
#pragma unroll
    for (int n = 0; n < 4; ++n) {
      v16bf bfv[4];
#pragma unroll
      for (int kk = 0; kk < 4; ++kk) {
        const unsigned short* p = &Kls[n * 16 + lm][kk * 32 + 16 * sel];
        bfv[kk] = frag16(p, p + 8);
      }
      v8f a = vzero;
#pragma unroll
      for (int kk = 0; kk < 4; ++kk)
        a = __builtin_amdgcn_wmma_f32_16x16x32_bf16(
            false, qf[kk], false, bfv[kk], (short)0, a, false, false);
      s[n] = a;
    }

    // ---- scale + causal mask on the diagonal block ----
    const bool diag = (j == qblk);
#pragma unroll
    for (int n = 0; n < 4; ++n)
#pragma unroll
      for (int r = 0; r < 8; ++r) {
        float v = s[n][r] * scale;
        if (diag) {
          const int ql = wave * 16 + r + 8 * sel;
          const int kl = n * 16 + lm;
          if (kl > ql) v = -1e30f;
        }
        s[n][r] = v;
      }

    // ---- online softmax per row (row = r + 8*sel; 16 lanes share a row) ----
#pragma unroll
    for (int r = 0; r < 8; ++r) {
      float mx = fmaxf(fmaxf(s[0][r], s[1][r]), fmaxf(s[2][r], s[3][r]));
#pragma unroll
      for (int off = 1; off < 16; off <<= 1)
        mx = fmaxf(mx, __shfl_xor(mx, off, 32));
      const float mnew = fmaxf(mrow[r], mx);
      const float f = __expf(mrow[r] - mnew);
      float ssum = 0.f;
#pragma unroll
      for (int n = 0; n < 4; ++n) {
        const float p = __expf(s[n][r] - mnew);
        s[n][r] = p;
        ssum += p;
      }
#pragma unroll
      for (int off = 1; off < 16; off <<= 1)
        ssum += __shfl_xor(ssum, off, 32);
      lrow[r] = lrow[r] * f + ssum;
      mrow[r] = mnew;
#pragma unroll
      for (int n2 = 0; n2 < 8; ++n2) o[n2][r] *= f;
    }

    // ---- restage P (C-layout f32 -> A-layout bf16) via wave-private LDS ----
#pragma unroll
    for (int n = 0; n < 4; ++n)
#pragma unroll
      for (int r = 0; r < 8; ++r)
        Pls[wave][r + 8 * sel][n * 16 + lm] = f32_to_bf16(s[n][r]);
    // Same-wave DS ops are in order; compiler inserts the dscnt wait.

    // ---- O += P * V: 2 K-steps; V fragments preloaded in groups of 4 ----
#pragma unroll
    for (int kk2 = 0; kk2 < 2; ++kk2) {
      const unsigned short* pp = &Pls[wave][lm][kk2 * 32 + 16 * sel];
      v16bf pf = frag16(pp, pp + 8);
#pragma unroll
      for (int g = 0; g < 2; ++g) {
        v16bf vf[4];
#pragma unroll
        for (int q = 0; q < 4; ++q) {
          const unsigned short* vp = &Vt[(g * 4 + q) * 16 + lm][kk2 * 32 + 16 * sel];
          vf[q] = frag16(vp, vp + 8);
        }
#pragma unroll
        for (int q = 0; q < 4; ++q)
          o[g * 4 + q] = __builtin_amdgcn_wmma_f32_16x16x32_bf16(
              false, pf, false, vf[q], (short)0, o[g * 4 + q], false, false);
      }
    }
    __syncthreads();
  }

  // ---- finalize: O /= l, write y as bf16 (head-interleaved [B,T,C]) ----
#pragma unroll
  for (int n2 = 0; n2 < 8; ++n2)
#pragma unroll
    for (int r = 0; r < 8; ++r) {
      const int row = qblk * 64 + wave * 16 + r + 8 * sel;
      const float v = o[n2][r] / lrow[r];
      Y[(size_t)(b * T + row) * C + h * D + n2 * 16 + lm] = f32_to_bf16(v);
    }
}

// ---------------------------------------------------------------------------
// Launch
// ---------------------------------------------------------------------------
extern "C" void kernel_launch(void* const* d_in, const int* in_sizes, int n_in,
                              void* d_out, int out_size, void* d_ws, size_t ws_size,
                              hipStream_t stream) {
  const float* x  = (const float*)d_in[0];   // [8,1024,2048]
  const float* wa = (const float*)d_in[1];   // [2048,6144]
  const float* wp = (const float*)d_in[2];   // [2048,2048]
  float* out = (float*)d_out;                // [8,1024,2048]

  const int Bb = 8, T = 1024, C = 2048;
  const size_t nx   = (size_t)Bb * T * C;        // 16,777,216
  const size_t nwa  = (size_t)C * 3 * C;         // 12,582,912
  const size_t nwp  = (size_t)C * C;             //  4,194,304
  const size_t nqkv = (size_t)Bb * T * 3 * C;    // 50,331,648
  const size_t ny   = nx;

  char* ws = (char*)d_ws;
  auto carve = [&](size_t elems) {
    unsigned short* p = (unsigned short*)ws;
    ws += ((elems * sizeof(unsigned short) + 255) / 256) * 256;
    return p;
  };
  unsigned short* xb   = carve(nx);
  unsigned short* wab  = carve(nwa);
  unsigned short* wpb  = carve(nwp);
  unsigned short* qkvb = carve(nqkv);
  unsigned short* yb   = carve(ny);

  cvt_f32_bf16<<<2048, 256, 0, stream>>>(x,  xb,  nx);
  cvt_f32_bf16<<<2048, 256, 0, stream>>>(wa, wab, nwa);
  cvt_f32_bf16<<<1024, 256, 0, stream>>>(wp, wpb, nwp);

  // qkv = x @ w_attn   [8192 x 6144], K = 2048
  gemm_bf16_wmma<unsigned short>
      <<<dim3(6144 / 128, 8192 / 128), 256, 0, stream>>>(xb, wab, qkvb,
                                                         8192, 6144, 2048);
  // flash attention -> y (bf16)
  attn_flash_wmma<<<dim3(T / 64, Bb * 16), 128, 0, stream>>>(qkvb, yb);

  // out = y @ w_proj   [8192 x 2048], K = 2048, f32 output
  gemm_bf16_wmma<float>
      <<<dim3(2048 / 128, 8192 / 128), 256, 0, stream>>>(yb, wpb, out,
                                                         8192, 2048, 2048);
}
